// BiLSTM_20736102105741
// MI455X (gfx1250) — compile-verified, matrix operand order checked by reading, not run on hardware
//
#include <hip/hip_runtime.h>
#include <hip/hip_bf16.h>

typedef __attribute__((ext_vector_type(16))) _Float16 v16h;
typedef __attribute__((ext_vector_type(8)))  _Float16 v8h;
typedef __attribute__((ext_vector_type(8)))  float    v8f;
typedef int v4i __attribute__((vector_size(16)));   // matches async-builtin param

#define B_   32
#define T_   512
#define I_   256
#define H_   512
#define O_   128
#define G_   2048           // 4*H
#define NWG  8              // workgroups per direction in the scan
#define HS   64             // H columns per workgroup (H_/NWG)
#define LDP  520            // padded LDS row stride (f16) to avoid bank conflicts

// ---------------------------------------------------------------------------
// Fragment loader for 16-bit WMMA A/B operands (16x32 / 32x16 tiles).
// Per ISA layout: lane holds row/col (lane&15); VGPRs 0..3 hold K = kb..kb+7,
// VGPRs 4..7 hold K = kb+16..kb+23, where kb = 8*(lane>>4) (+ loop k offset).
// ---------------------------------------------------------------------------
__device__ __forceinline__ v16h frag_k(const _Float16* rowp, int k) {
  v8h lo = *(const v8h*)(rowp + k);
  v8h hi = *(const v8h*)(rowp + k + 16);
  v16h f;
#pragma unroll
  for (int i = 0; i < 8; ++i) { f[i] = lo[i]; f[i + 8] = hi[i]; }
  return f;
}

__device__ __forceinline__ v8f wmma_f16(v16h a, v16h b, v8f c) {
  return __builtin_amdgcn_wmma_f32_16x16x32_f16(false, a, false, b,
                                                (short)0, c, false, false);
}

// 16-byte async global->LDS copy (CDNA5 async engine), with safe fallback.
__device__ __forceinline__ void async_copy16(const _Float16* g, _Float16* l) {
#if __has_builtin(__builtin_amdgcn_global_load_async_to_lds_b128)
  __builtin_amdgcn_global_load_async_to_lds_b128((v4i*)g, (v4i*)l, 0, 0);
#else
  *(uint4*)l = *(const uint4*)g;
#endif
}

__device__ __forceinline__ void wait_async0() {
#if __has_builtin(__builtin_amdgcn_s_wait_asynccnt)
  __builtin_amdgcn_s_wait_asynccnt(0);
#else
  asm volatile("s_wait_asynccnt 0" ::: "memory");
#endif
}

__device__ __forceinline__ float sigf(float x) {
  return 1.0f / (1.0f + __expf(-x));
}
__device__ __forceinline__ float tanha(float x) {
  x = fminf(15.0f, fmaxf(-15.0f, x));
  float e = __expf(2.0f * x);
  return (e - 1.0f) / (e + 1.0f);
}

// ---------------------------------------------------------------------------
// Prep kernels
// ---------------------------------------------------------------------------
__global__ void prep_f32_to_f16(const float* __restrict__ src,
                                _Float16* __restrict__ dst, int n) {
  int i = blockIdx.x * blockDim.x + threadIdx.x;
  if (i < n) dst[i] = (_Float16)src[i];
}

// Split the 4 gate matrices [H, I+H] into Wx f16 [4H, I] and Wh f16 [4H, H].
__global__ void prep_weights(const float* __restrict__ w0, const float* __restrict__ w1,
                             const float* __restrict__ w2, const float* __restrict__ w3,
                             _Float16* __restrict__ wx, _Float16* __restrict__ wh) {
  int idx = blockIdx.x * blockDim.x + threadIdx.x;
  const int n = 4 * H_ * (I_ + H_);
  if (idx >= n) return;
  int g   = idx / (H_ * (I_ + H_));
  int rem = idx % (H_ * (I_ + H_));
  int r = rem / (I_ + H_);
  int c = rem % (I_ + H_);
  const float* w = (g == 0) ? w0 : (g == 1) ? w1 : (g == 2) ? w2 : w3;
  float v = w[(size_t)r * (I_ + H_) + c];
  if (c < I_) wx[((size_t)g * H_ + r) * I_ + c] = (_Float16)v;
  else        wh[((size_t)g * H_ + r) * H_ + (c - I_)] = (_Float16)v;
}

__global__ void prep_bias(const float* __restrict__ b0, const float* __restrict__ b1,
                          const float* __restrict__ b2, const float* __restrict__ b3,
                          float* __restrict__ bias) {
  int idx = blockIdx.x * blockDim.x + threadIdx.x;
  if (idx >= G_) return;
  int g = idx >> 9, j = idx & (H_ - 1);
  const float* b = (g == 0) ? b0 : (g == 1) ? b1 : (g == 2) ? b2 : b3;
  bias[idx] = b[j];
}

// slot 0 of each direction's hidden ring + sync counters.
__global__ void prep_h0(const float* __restrict__ bh0,
                        _Float16* __restrict__ hseqF, _Float16* __restrict__ hseqB,
                        int* __restrict__ counters) {
  int idx = blockIdx.x * blockDim.x + threadIdx.x;
  if (idx < B_ * H_) {
    hseqF[idx] = (_Float16)0.0f;
    hseqB[idx] = (_Float16)bh0[idx & (H_ - 1)];
  }
  if (idx < 2) counters[idx * 64] = 0;
}

// ---------------------------------------------------------------------------
// xproj GEMM: xp[t*B+b][g] = x[b,t,:] . Wx[g,:] + bias[g]
// M = T*B = 16384 (m = t*32+b), N = 2048, K = 256. 1 wave = 16x64 strip.
// ---------------------------------------------------------------------------
__global__ void xproj_gemm(const _Float16* __restrict__ xf,
                           const _Float16* __restrict__ wx,
                           const float* __restrict__ bias,
                           float* __restrict__ xp) {
  const int lane  = threadIdx.x & 31;
  const int wv    = threadIdx.x >> 5;
  const int widx  = blockIdx.x * 8 + wv;   // 0..32767
  const int mtile = widx >> 5;             // 0..1023
  const int ng    = widx & 31;             // 64-col group
  const int lrow  = lane & 15;
  const int lhalf = lane >> 4;
  const int kq    = lhalf * 8;

  int m = mtile * 16 + lrow;               // A-frag row
  int t = m >> 5, b = m & 31;
  const _Float16* arow = xf + ((size_t)b * T_ + t) * I_;

  const int n0 = ng * 64;
  v8f acc[4];
  const _Float16* brow[4];
#pragma unroll
  for (int j = 0; j < 4; ++j) {
    float bv = bias[n0 + j * 16 + lrow];
#pragma unroll
    for (int r = 0; r < 8; ++r) acc[j][r] = bv;
    brow[j] = wx + (size_t)(n0 + j * 16 + lrow) * I_;
  }

#pragma unroll
  for (int k = 0; k < I_; k += 32) {
    v16h a = frag_k(arow, k + kq);
#pragma unroll
    for (int j = 0; j < 4; ++j) {
      v16h bf = frag_k(brow[j], k + kq);
      acc[j] = wmma_f16(a, bf, acc[j]);
    }
  }

#pragma unroll
  for (int j = 0; j < 4; ++j)
#pragma unroll
    for (int r = 0; r < 8; ++r) {
      int mm = mtile * 16 + lhalf * 8 + r;
      xp[(size_t)mm * G_ + n0 + j * 16 + lrow] = acc[j][r];
    }
}

// ---------------------------------------------------------------------------
// Persistent bidirectional LSTM scan.
// Grid = 16 blocks: dir = blk>>3, wg = blk&7 owns h-cols [wg*64, wg*64+64).
// LDS: Wh slice (256 rows, padded) resident for the whole scan + a per-step
// h_prev staging buffer, both filled with GLOBAL_LOAD_ASYNC_TO_LDS_B128.
// Cell state c lives in registers (one v8f per wave).
// h ring: slot s = h before step s; step s writes slot s+1. Grid sync via
// monotonic atomic counter per direction.
// ---------------------------------------------------------------------------
__global__ void lstm_scan(const _Float16* __restrict__ whf,
                          const float* __restrict__ xp,
                          const float* __restrict__ bc0,
                          _Float16* __restrict__ hseqF,
                          _Float16* __restrict__ hseqB,
                          int* __restrict__ counters) {
  extern __shared__ _Float16 lds[];        // [256][LDP] Wh + [32][LDP] h
  _Float16* whs  = lds;
  _Float16* hbuf = lds + 256 * LDP;

  const int dir = blockIdx.x >> 3;
  const int wg  = blockIdx.x & 7;
  const int h0  = wg * HS;
  _Float16* hseq = dir ? hseqB : hseqF;
  int* cnt = counters + dir * 64;

  const int lane  = threadIdx.x & 31;
  const int wv    = threadIdx.x >> 5;
  const int mtile = wv & 1;                // batch rows 0-15 / 16-31
  const int q     = wv >> 1;               // 16-col quarter of the h-slice
  const int lrow  = lane & 15;
  const int lhalf = lane >> 4;
  const int kq    = lhalf * 8;

  // Stage Wh slice into LDS (async): local row nl = gate*64 + rem  <->
  // global row gate*512 + h0 + rem; 16-byte transfers, padded LDS rows.
  for (int idx = threadIdx.x; idx < 256 * 64; idx += 256) {
    int nl = idx >> 6, kc = idx & 63;
    int gate = nl >> 6, rem = nl & 63;
    int grow = gate * H_ + h0 + rem;
    async_copy16(whf + (size_t)grow * H_ + kc * 8, whs + (size_t)nl * LDP + kc * 8);
  }
  wait_async0();
  __syncthreads();

  // Step-invariant B row pointers (LDS) and A row offset (LDS h buffer).
  const _Float16* brow[4];
#pragma unroll
  for (int j = 0; j < 4; ++j)
    brow[j] = whs + (size_t)(j * HS + q * 16 + lrow) * LDP;
  const _Float16* arow = hbuf + (size_t)(mtile * 16 + lrow) * LDP;
  const int mycol = h0 + q * 16 + lrow;    // this lane's h column

  // Cell state in registers: element (r) <-> batch m = mtile*16+lhalf*8+r.
  v8f cst;
  {
    float cv = dir ? bc0[mycol] : 0.0f;
#pragma unroll
    for (int r = 0; r < 8; ++r) cst[r] = cv;
  }

  for (int s = 0; s < T_; ++s) {
    const int t = dir ? (T_ - 1 - s) : s;
    const float* xrow = xp + (size_t)t * B_ * G_;

    // Kick off async staging of h_prev (ring slot s) into LDS.
    const _Float16* hsrc = hseq + (size_t)s * (B_ * H_);
    for (int idx = threadIdx.x; idx < B_ * 64; idx += 256) {
      int row = idx >> 6, kc = idx & 63;
      async_copy16(hsrc + (size_t)row * H_ + kc * 8,
                   hbuf + (size_t)row * LDP + kc * 8);
    }

    // Overlap: accumulators initialized from the precomputed input projection.
    v8f acc[4];
#pragma unroll
    for (int j = 0; j < 4; ++j) {
      const int col = j * H_ + mycol;
#pragma unroll
      for (int r = 0; r < 8; ++r) {
        int m = mtile * 16 + lhalf * 8 + r;
        acc[j][r] = xrow[(size_t)m * G_ + col];
      }
    }

    // Prefetch next step's xproj rows into cache.
    if (s + 1 < T_) {
      int tn = dir ? (T_ - 2 - s) : (s + 1);
      __builtin_prefetch(xp + ((size_t)tn * B_ + lane) * G_ + h0, 0, 1);
    }

    wait_async0();
    __syncthreads();

    // h_prev @ Wh_slice^T  (A and B both from LDS).
#pragma unroll 4
    for (int k = 0; k < H_; k += 32) {
      v16h a = frag_k(arow, k + kq);
#pragma unroll
      for (int j = 0; j < 4; ++j) {
        v16h bf = frag_k(brow[j], k + kq);
        acc[j] = wmma_f16(a, bf, acc[j]);
      }
    }

    // Gates + state update, write h into ring slot s+1.
    _Float16* hout = hseq + (size_t)(s + 1) * (B_ * H_);
#pragma unroll
    for (int r = 0; r < 8; ++r) {
      float fg = sigf(acc[0][r]);
      float ig = sigf(acc[1][r]);
      float og = sigf(acc[2][r]);
      float ch = tanha(acc[3][r]);
      float cc = fg * cst[r] + ig * ch;
      cst[r] = cc;
      float hh = og * tanha(cc);
      int m = mtile * 16 + lhalf * 8 + r;
      hout[(size_t)m * H_ + mycol] = (_Float16)hh;
    }

    // Grid sync within this direction (monotonic counter, graph-safe).
    __threadfence();
    __syncthreads();
    if (threadIdx.x == 0) {
      __hip_atomic_fetch_add(cnt, 1, __ATOMIC_RELEASE, __HIP_MEMORY_SCOPE_AGENT);
      const int tgt = NWG * (s + 1);
      while (__hip_atomic_load(cnt, __ATOMIC_ACQUIRE, __HIP_MEMORY_SCOPE_AGENT) < tgt)
        __builtin_amdgcn_s_sleep(2);
    }
    __syncthreads();
    __threadfence();
  }
}

// ---------------------------------------------------------------------------
// Output GEMM: out[b,t,:] = [hf(t,b); hb(t,b)] @ out_w^T + out_b
// m = b*T + t, K = 1024 (hf then hb), N = 128. 1 wave = 16 x 128 strip.
// ---------------------------------------------------------------------------
__global__ void out_gemm(const _Float16* __restrict__ hseqF,
                         const _Float16* __restrict__ hseqB,
                         const _Float16* __restrict__ owf,
                         const float* __restrict__ outb,
                         float* __restrict__ out) {
  const int lane  = threadIdx.x & 31;
  const int wv    = threadIdx.x >> 5;
  const int mtile = blockIdx.x * 8 + wv;   // 0..1023
  const int lrow  = lane & 15;
  const int lhalf = lane >> 4;
  const int kq    = lhalf * 8;

  int m = mtile * 16 + lrow;
  int b = m >> 9, t = m & (T_ - 1);
  // hf[t] = slotF[t+1], hb[t] = slotB[T-t]
  const _Float16* afrow = hseqF + ((size_t)(t + 1) * B_ + b) * H_;
  const _Float16* abrow = hseqB + ((size_t)(T_ - t) * B_ + b) * H_;

  v8f acc[8];
  const _Float16* brow[8];
#pragma unroll
  for (int j = 0; j < 8; ++j) {
    float bv = outb[j * 16 + lrow];
#pragma unroll
    for (int r = 0; r < 8; ++r) acc[j][r] = bv;
    brow[j] = owf + (size_t)(j * 16 + lrow) * (2 * H_);
  }

#pragma unroll 4
  for (int k = 0; k < H_; k += 32) {
    v16h a = frag_k(afrow, k + kq);
#pragma unroll
    for (int j = 0; j < 8; ++j) {
      v16h bf = frag_k(brow[j], k + kq);
      acc[j] = wmma_f16(a, bf, acc[j]);
    }
  }
#pragma unroll 4
  for (int k = 0; k < H_; k += 32) {
    v16h a = frag_k(abrow, k + kq);
#pragma unroll
    for (int j = 0; j < 8; ++j) {
      v16h bf = frag_k(brow[j], H_ + k + kq);
      acc[j] = wmma_f16(a, bf, acc[j]);
    }
  }

#pragma unroll
  for (int j = 0; j < 8; ++j)
#pragma unroll
    for (int r = 0; r < 8; ++r) {
      int mm = mtile * 16 + lhalf * 8 + r;
      out[(size_t)mm * O_ + j * 16 + lrow] = acc[j][r];
    }
}

// ---------------------------------------------------------------------------
// Workspace layout (bytes, all 256-aligned)
// ---------------------------------------------------------------------------
#define OFF_XF    ((size_t)0)                              //  8,388,608  x f16
#define OFF_WX    (OFF_XF   + (size_t)B_*T_*I_*2)          //  1,048,576  Wx f16
#define OFF_WH    (OFF_WX   + (size_t)G_*I_*2)             //  2,097,152  Wh f16
#define OFF_OW    (OFF_WH   + (size_t)G_*H_*2)             //    262,144  out_w f16
#define OFF_BIAS  (OFF_OW   + (size_t)O_*2*H_*2)           //      8,192  bias f32
#define OFF_CNT   (OFF_BIAS + (size_t)G_*4)                //        512  counters
#define OFF_HF    (OFF_CNT  + (size_t)512)                 // 16,809,984  hseqF
#define OFF_HB    (OFF_HF   + (size_t)(T_+1)*B_*H_*2)      // 16,809,984  hseqB
#define OFF_XP    (OFF_HB   + (size_t)(T_+1)*B_*H_*2)      // 134,217,728 xproj f32

extern "C" void kernel_launch(void* const* d_in, const int* in_sizes, int n_in,
                              void* d_out, int out_size, void* d_ws, size_t ws_size,
                              hipStream_t stream) {
  const float* x    = (const float*)d_in[0];
  const float* Wf_w = (const float*)d_in[1];
  const float* Wf_b = (const float*)d_in[2];
  const float* Wi_w = (const float*)d_in[3];
  const float* Wi_b = (const float*)d_in[4];
  const float* Wo_w = (const float*)d_in[5];
  const float* Wo_b = (const float*)d_in[6];
  const float* Wc_w = (const float*)d_in[7];
  const float* Wc_b = (const float*)d_in[8];
  const float* ow   = (const float*)d_in[9];
  const float* ob   = (const float*)d_in[10];
  const float* bh0  = (const float*)d_in[11];
  const float* bc0  = (const float*)d_in[12];

  char* ws = (char*)d_ws;
  _Float16* xf    = (_Float16*)(ws + OFF_XF);
  _Float16* wx    = (_Float16*)(ws + OFF_WX);
  _Float16* wh    = (_Float16*)(ws + OFF_WH);
  _Float16* owf   = (_Float16*)(ws + OFF_OW);
  float*    bias  = (float*)   (ws + OFF_BIAS);
  int*      cnt   = (int*)     (ws + OFF_CNT);
  _Float16* hseqF = (_Float16*)(ws + OFF_HF);
  _Float16* hseqB = (_Float16*)(ws + OFF_HB);
  float*    xp    = (float*)   (ws + OFF_XP);

  const int nX = B_ * T_ * I_;               // 4,194,304
  const int nW = 4 * H_ * (I_ + H_);         // 1,572,864
  const int nO = O_ * 2 * H_;                // 131,072

  prep_f32_to_f16<<<(nX + 255) / 256, 256, 0, stream>>>(x, xf, nX);
  prep_weights<<<(nW + 255) / 256, 256, 0, stream>>>(Wf_w, Wi_w, Wo_w, Wc_w, wx, wh);
  prep_bias<<<(G_ + 255) / 256, 256, 0, stream>>>(Wf_b, Wi_b, Wo_b, Wc_b, bias);
  prep_f32_to_f16<<<(nO + 255) / 256, 256, 0, stream>>>(ow, owf, nO);
  prep_h0<<<(B_ * H_ + 255) / 256, 256, 0, stream>>>(bh0, hseqF, hseqB, cnt);

  // 17.2 GFLOP input projection: 32768 waves, 1 wave = 16x64 tile strip.
  xproj_gemm<<<4096, 256, 0, stream>>>(xf, wx, bias, xp);

  // Persistent recurrent scan: 2 dirs x 8 WGs, ~292.5 KB LDS each
  // (padded Wh slice resident + per-step h staging buffer).
  lstm_scan<<<2 * NWG, 256, (256 + B_) * LDP * (int)sizeof(_Float16), stream>>>(
      wh, xp, bc0, hseqF, hseqB, cnt);

  // Output projection: 1024 M-tiles, 8 waves/block.
  out_gemm<<<128, 256, 0, stream>>>(hseqF, hseqB, owf, ob, (float*)d_out);
}
